// QuadraticModel_58351425683935
// MI455X (gfx1250) — matrix operand order repaired and checked by reading, not compiled, not run on hardware
//
#include <hip/hip_runtime.h>

#define DIM    4096
#define BATCH  8192
#define BM     128     // batch rows per workgroup
#define BN     128     // Y columns (k) per chunk
#define BK     32      // d-step per WMMA sweep
#define NSTEP  (DIM / BK)
#define NCHUNK 4       // k-chunks per workgroup -> 512 cols per n-group
#define NGROUP 8       // gridDim.y ; NGROUP*NCHUNK*BN == DIM
#define BPITCH 40      // bf16 elems per padded LDS row (32 data + 8 pad) = 80 B

typedef __attribute__((ext_vector_type(16))) __bf16 v16bf;
typedef __attribute__((ext_vector_type(8)))  __bf16 v8bf;
typedef __attribute__((ext_vector_type(8)))  float  v8f;

// ---------------------------------------------------------------------------
// One-time fp32 -> (bf16 hi, bf16 lo) split pass (bandwidth-trivial).
// Each thread handles 8 consecutive elements.
// ---------------------------------------------------------------------------
__global__ __launch_bounds__(256) void split_kernel(
    const float* __restrict__ src, __bf16* __restrict__ hi,
    __bf16* __restrict__ lo)
{
  const size_t base = ((size_t)blockIdx.x * 256 + threadIdx.x) * 8;
  float f[8];
  *(float4*)&f[0] = *(const float4*)(src + base);
  *(float4*)&f[4] = *(const float4*)(src + base + 4);
  v8bf h, l;
#pragma unroll
  for (int i = 0; i < 8; ++i) {
    __bf16 hv = (__bf16)f[i];
    h[i] = hv;
    l[i] = (__bf16)(f[i] - (float)hv);
  }
  *(v8bf*)(hi + base) = h;
  *(v8bf*)(lo + base) = l;
}

// ---------------------------------------------------------------------------
// Fast path: Y = x @ sqrtQ^T from prebuilt bf16 hi/lo planes.
// No conversions in the loop: global b128 -> VGPR -> ds b128, double-buffered
// LDS, one barrier per step, 24 v_wmma per step per wave.
// ---------------------------------------------------------------------------
__global__ __launch_bounds__(256) void quad_gemm_planes_kernel(
    const __bf16* __restrict__ xh, const __bf16* __restrict__ xl,
    const __bf16* __restrict__ qh, const __bf16* __restrict__ ql,
    float* __restrict__ ws)
{
  __shared__ __bf16 sAh[2][BM][BPITCH];
  __shared__ __bf16 sAl[2][BM][BPITCH];
  __shared__ __bf16 sBh[2][BM][BPITCH];
  __shared__ __bf16 sBl[2][BM][BPITCH];
  __shared__ float  rowsum[2][BM];

  const int t    = threadIdx.x;
  const int lane = t & 31;
  const int wave = t >> 5;
  const int wm   = wave & 3;
  const int wn   = wave >> 2;
  const int half = lane >> 4;
  const int lsub = lane & 15;

  const int m0 = blockIdx.x * BM;

  const int frow = t >> 1;
  const int fdh  = (t & 1) * 16;                 // element offset in row
  const __bf16* Ah = xh + (size_t)(m0 + frow) * DIM + fdh;
  const __bf16* Al = xl + (size_t)(m0 + frow) * DIM + fdh;

  float qacc[16];
#pragma unroll
  for (int i = 0; i < 16; ++i) qacc[i] = 0.f;

  for (int nc = 0; nc < NCHUNK; ++nc) {
    const int k0 = blockIdx.y * (NCHUNK * BN) + nc * BN;
    const __bf16* Bh = qh + (size_t)(k0 + frow) * DIM + fdh;
    const __bf16* Bl = ql + (size_t)(k0 + frow) * DIM + fdh;

    v8f acc[2][4];
#pragma unroll
    for (int mt = 0; mt < 2; ++mt)
#pragma unroll
      for (int nt = 0; nt < 4; ++nt)
#pragma unroll
        for (int i = 0; i < 8; ++i) acc[mt][nt][i] = 0.f;

    v8bf rah0, rah1, ral0, ral1, rbh0, rbh1, rbl0, rbl1;

    // prologue: stage step 0 into buffer 0
    rah0 = *(const v8bf*)(Ah);       rah1 = *(const v8bf*)(Ah + 8);
    ral0 = *(const v8bf*)(Al);       ral1 = *(const v8bf*)(Al + 8);
    rbh0 = *(const v8bf*)(Bh);       rbh1 = *(const v8bf*)(Bh + 8);
    rbl0 = *(const v8bf*)(Bl);       rbl1 = *(const v8bf*)(Bl + 8);
    *(v8bf*)&sAh[0][frow][fdh] = rah0;  *(v8bf*)&sAh[0][frow][fdh + 8] = rah1;
    *(v8bf*)&sAl[0][frow][fdh] = ral0;  *(v8bf*)&sAl[0][frow][fdh + 8] = ral1;
    *(v8bf*)&sBh[0][frow][fdh] = rbh0;  *(v8bf*)&sBh[0][frow][fdh + 8] = rbh1;
    *(v8bf*)&sBl[0][frow][fdh] = rbl0;  *(v8bf*)&sBl[0][frow][fdh + 8] = rbl1;
    __syncthreads();

    for (int step = 0; step < NSTEP; ++step) {
      const int p = step & 1;

      // issue global loads for step+1 early; latency hides under the WMMAs
      if (step + 1 < NSTEP) {
        const size_t off = (size_t)(step + 1) * BK;
        rah0 = *(const v8bf*)(Ah + off);      rah1 = *(const v8bf*)(Ah + off + 8);
        ral0 = *(const v8bf*)(Al + off);      ral1 = *(const v8bf*)(Al + off + 8);
        rbh0 = *(const v8bf*)(Bh + off);      rbh1 = *(const v8bf*)(Bh + off + 8);
        rbl0 = *(const v8bf*)(Bl + off);      rbl1 = *(const v8bf*)(Bl + off + 8);
      }

      // ---- load fragments from LDS buffer p ----
      v16bf ah[2], al[2], bh[4], bl[4];
#pragma unroll
      for (int mt = 0; mt < 2; ++mt) {
        const __bf16* ph = &sAh[p][wm * 32 + mt * 16 + lsub][0];
        const __bf16* pl = &sAl[p][wm * 32 + mt * 16 + lsub][0];
        v8bf h0 = *(const v8bf*)(ph + half * 8);
        v8bf h1 = *(const v8bf*)(ph + 16 + half * 8);
        v8bf l0 = *(const v8bf*)(pl + half * 8);
        v8bf l1 = *(const v8bf*)(pl + 16 + half * 8);
        ah[mt] = __builtin_shufflevector(h0, h1, 0,1,2,3,4,5,6,7,8,9,10,11,12,13,14,15);
        al[mt] = __builtin_shufflevector(l0, l1, 0,1,2,3,4,5,6,7,8,9,10,11,12,13,14,15);
      }
#pragma unroll
      for (int nt = 0; nt < 4; ++nt) {
        const __bf16* ph = &sBh[p][wn * 64 + nt * 16 + lsub][0];
        const __bf16* pl = &sBl[p][wn * 64 + nt * 16 + lsub][0];
        v8bf h0 = *(const v8bf*)(ph + half * 8);
        v8bf h1 = *(const v8bf*)(ph + 16 + half * 8);
        v8bf l0 = *(const v8bf*)(pl + half * 8);
        v8bf l1 = *(const v8bf*)(pl + 16 + half * 8);
        bh[nt] = __builtin_shufflevector(h0, h1, 0,1,2,3,4,5,6,7,8,9,10,11,12,13,14,15);
        bl[nt] = __builtin_shufflevector(l0, l1, 0,1,2,3,4,5,6,7,8,9,10,11,12,13,14,15);
      }

      // ---- split-precision MMA: hi*hi + hi*lo + lo*hi ----
#pragma unroll
      for (int mt = 0; mt < 2; ++mt)
#pragma unroll
        for (int nt = 0; nt < 4; ++nt) {
          acc[mt][nt] = __builtin_amdgcn_wmma_f32_16x16x32_bf16(
              false, ah[mt], false, bh[nt], (short)0, acc[mt][nt], false, false);
          acc[mt][nt] = __builtin_amdgcn_wmma_f32_16x16x32_bf16(
              false, ah[mt], false, bl[nt], (short)0, acc[mt][nt], false, false);
          acc[mt][nt] = __builtin_amdgcn_wmma_f32_16x16x32_bf16(
              false, al[mt], false, bh[nt], (short)0, acc[mt][nt], false, false);
        }

      // ---- store step+1 into the other buffer (overlaps WMMAs) ----
      if (step + 1 < NSTEP) {
        const int pn = p ^ 1;
        *(v8bf*)&sAh[pn][frow][fdh] = rah0;  *(v8bf*)&sAh[pn][frow][fdh + 8] = rah1;
        *(v8bf*)&sAl[pn][frow][fdh] = ral0;  *(v8bf*)&sAl[pn][frow][fdh + 8] = ral1;
        *(v8bf*)&sBh[pn][frow][fdh] = rbh0;  *(v8bf*)&sBh[pn][frow][fdh + 8] = rbh1;
        *(v8bf*)&sBl[pn][frow][fdh] = rbl0;  *(v8bf*)&sBl[pn][frow][fdh + 8] = rbl1;
      }

      __syncthreads();
    }

    // square-accumulate this chunk's Y tile
#pragma unroll
    for (int mt = 0; mt < 2; ++mt)
#pragma unroll
      for (int v = 0; v < 8; ++v) {
        float s = 0.f;
#pragma unroll
        for (int nt = 0; nt < 4; ++nt) { float y = acc[mt][nt][v]; s += y * y; }
        qacc[mt * 8 + v] += s;
      }
  }

#pragma unroll
  for (int i = 0; i < 16; ++i) {
    float s = qacc[i];
    s += __shfl_xor(s, 8, 32);
    s += __shfl_xor(s, 4, 32);
    s += __shfl_xor(s, 2, 32);
    s += __shfl_xor(s, 1, 32);
    qacc[i] = s;
  }
  if (lsub == 0) {
#pragma unroll
    for (int mt = 0; mt < 2; ++mt)
#pragma unroll
      for (int v = 0; v < 8; ++v)
        rowsum[wn][wm * 32 + mt * 16 + half * 8 + v] = qacc[mt * 8 + v];
  }
  __syncthreads();
  if (t < BM)
    ws[(size_t)blockIdx.y * BATCH + m0 + t] = rowsum[0][t] + rowsum[1][t];
}

// ---------------------------------------------------------------------------
// Fallback path (small workspace): convert fp32 -> hi/lo in-loop.
// ---------------------------------------------------------------------------
__global__ __launch_bounds__(256) void quad_gemm_kernel(
    const float* __restrict__ x, const float* __restrict__ sq,
    float* __restrict__ ws)
{
  __shared__ __bf16 sAh[2][BM][BPITCH];
  __shared__ __bf16 sAl[2][BM][BPITCH];
  __shared__ __bf16 sBh[2][BM][BPITCH];
  __shared__ __bf16 sBl[2][BM][BPITCH];
  __shared__ float  rowsum[2][BM];

  const int t    = threadIdx.x;
  const int lane = t & 31;
  const int wave = t >> 5;
  const int wm   = wave & 3;
  const int wn   = wave >> 2;
  const int half = lane >> 4;
  const int lsub = lane & 15;

  const int m0 = blockIdx.x * BM;
  const int frow = t >> 1;
  const int fdh  = (t & 1) * 16;
  const float* Abase = x + (size_t)(m0 + frow) * DIM + fdh;

  float qacc[16];
#pragma unroll
  for (int i = 0; i < 16; ++i) qacc[i] = 0.f;

  for (int nc = 0; nc < NCHUNK; ++nc) {
    const int k0 = blockIdx.y * (NCHUNK * BN) + nc * BN;
    const float* Bbase = sq + (size_t)(k0 + frow) * DIM + fdh;

    v8f acc[2][4];
#pragma unroll
    for (int mt = 0; mt < 2; ++mt)
#pragma unroll
      for (int nt = 0; nt < 4; ++nt)
#pragma unroll
        for (int i = 0; i < 8; ++i) acc[mt][nt][i] = 0.f;

    float4 ra[4], rb[4];
#pragma unroll
    for (int i = 0; i < 4; ++i) {
      ra[i] = *(const float4*)(Abase + i * 4);
      rb[i] = *(const float4*)(Bbase + i * 4);
    }
    {
      float fa[16], fb[16];
#pragma unroll
      for (int i = 0; i < 4; ++i) {
        fa[4*i+0] = ra[i].x; fa[4*i+1] = ra[i].y;
        fa[4*i+2] = ra[i].z; fa[4*i+3] = ra[i].w;
        fb[4*i+0] = rb[i].x; fb[4*i+1] = rb[i].y;
        fb[4*i+2] = rb[i].z; fb[4*i+3] = rb[i].w;
      }
      v8bf ah0, ah1, al0, al1, bh0, bh1, bl0, bl1;
#pragma unroll
      for (int i = 0; i < 8; ++i) {
        __bf16 h;
        h = (__bf16)fa[i];     ah0[i] = h; al0[i] = (__bf16)(fa[i]   - (float)h);
        h = (__bf16)fa[8+i];   ah1[i] = h; al1[i] = (__bf16)(fa[8+i] - (float)h);
        h = (__bf16)fb[i];     bh0[i] = h; bl0[i] = (__bf16)(fb[i]   - (float)h);
        h = (__bf16)fb[8+i];   bh1[i] = h; bl1[i] = (__bf16)(fb[8+i] - (float)h);
      }
      *(v8bf*)&sAh[0][frow][fdh]     = ah0;  *(v8bf*)&sAh[0][frow][fdh + 8] = ah1;
      *(v8bf*)&sAl[0][frow][fdh]     = al0;  *(v8bf*)&sAl[0][frow][fdh + 8] = al1;
      *(v8bf*)&sBh[0][frow][fdh]     = bh0;  *(v8bf*)&sBh[0][frow][fdh + 8] = bh1;
      *(v8bf*)&sBl[0][frow][fdh]     = bl0;  *(v8bf*)&sBl[0][frow][fdh + 8] = bl1;
    }
    __syncthreads();

    for (int step = 0; step < NSTEP; ++step) {
      const int p = step & 1;
      if (step + 1 < NSTEP) {
        const float* pa = Abase + (size_t)(step + 1) * BK;
        const float* pb = Bbase + (size_t)(step + 1) * BK;
#pragma unroll
        for (int i = 0; i < 4; ++i) {
          ra[i] = *(const float4*)(pa + i * 4);
          rb[i] = *(const float4*)(pb + i * 4);
        }
      }

      v16bf ah[2], al[2], bh[4], bl[4];
#pragma unroll
      for (int mt = 0; mt < 2; ++mt) {
        const __bf16* ph = &sAh[p][wm * 32 + mt * 16 + lsub][0];
        const __bf16* pl = &sAl[p][wm * 32 + mt * 16 + lsub][0];
        v8bf h0 = *(const v8bf*)(ph + half * 8);
        v8bf h1 = *(const v8bf*)(ph + 16 + half * 8);
        v8bf l0 = *(const v8bf*)(pl + half * 8);
        v8bf l1 = *(const v8bf*)(pl + 16 + half * 8);
        ah[mt] = __builtin_shufflevector(h0, h1, 0,1,2,3,4,5,6,7,8,9,10,11,12,13,14,15);
        al[mt] = __builtin_shufflevector(l0, l1, 0,1,2,3,4,5,6,7,8,9,10,11,12,13,14,15);
      }
#pragma unroll
      for (int nt = 0; nt < 4; ++nt) {
        const __bf16* ph = &sBh[p][wn * 64 + nt * 16 + lsub][0];
        const __bf16* pl = &sBl[p][wn * 64 + nt * 16 + lsub][0];
        v8bf h0 = *(const v8bf*)(ph + half * 8);
        v8bf h1 = *(const v8bf*)(ph + 16 + half * 8);
        v8bf l0 = *(const v8bf*)(pl + half * 8);
        v8bf l1 = *(const v8bf*)(pl + 16 + half * 8);
        bh[nt] = __builtin_shufflevector(h0, h1, 0,1,2,3,4,5,6,7,8,9,10,11,12,13,14,15);
        bl[nt] = __builtin_shufflevector(l0, l1, 0,1,2,3,4,5,6,7,8,9,10,11,12,13,14,15);
      }

#pragma unroll
      for (int mt = 0; mt < 2; ++mt)
#pragma unroll
        for (int nt = 0; nt < 4; ++nt) {
          acc[mt][nt] = __builtin_amdgcn_wmma_f32_16x16x32_bf16(
              false, ah[mt], false, bh[nt], (short)0, acc[mt][nt], false, false);
          acc[mt][nt] = __builtin_amdgcn_wmma_f32_16x16x32_bf16(
              false, ah[mt], false, bl[nt], (short)0, acc[mt][nt], false, false);
          acc[mt][nt] = __builtin_amdgcn_wmma_f32_16x16x32_bf16(
              false, al[mt], false, bh[nt], (short)0, acc[mt][nt], false, false);
        }

      if (step + 1 < NSTEP) {
        const int pn = p ^ 1;
        float fa[16], fb[16];
#pragma unroll
        for (int i = 0; i < 4; ++i) {
          fa[4*i+0] = ra[i].x; fa[4*i+1] = ra[i].y;
          fa[4*i+2] = ra[i].z; fa[4*i+3] = ra[i].w;
          fb[4*i+0] = rb[i].x; fb[4*i+1] = rb[i].y;
          fb[4*i+2] = rb[i].z; fb[4*i+3] = rb[i].w;
        }
        v8bf ah0, ah1, al0, al1, bh0, bh1, bl0, bl1;
#pragma unroll
        for (int i = 0; i < 8; ++i) {
          __bf16 h;
          h = (__bf16)fa[i];     ah0[i] = h; al0[i] = (__bf16)(fa[i]   - (float)h);
          h = (__bf16)fa[8+i];   ah1[i] = h; al1[i] = (__bf16)(fa[8+i] - (float)h);
          h = (__bf16)fb[i];     bh0[i] = h; bl0[i] = (__bf16)(fb[i]   - (float)h);
          h = (__bf16)fb[8+i];   bh1[i] = h; bl1[i] = (__bf16)(fb[8+i] - (float)h);
        }
        *(v8bf*)&sAh[pn][frow][fdh]     = ah0;  *(v8bf*)&sAh[pn][frow][fdh + 8] = ah1;
        *(v8bf*)&sAl[pn][frow][fdh]     = al0;  *(v8bf*)&sAl[pn][frow][fdh + 8] = al1;
        *(v8bf*)&sBh[pn][frow][fdh]     = bh0;  *(v8bf*)&sBh[pn][frow][fdh + 8] = bh1;
        *(v8bf*)&sBl[pn][frow][fdh]     = bl0;  *(v8bf*)&sBl[pn][frow][fdh + 8] = bl1;
      }
      __syncthreads();
    }

#pragma unroll
    for (int mt = 0; mt < 2; ++mt)
#pragma unroll
      for (int v = 0; v < 8; ++v) {
        float s = 0.f;
#pragma unroll
        for (int nt = 0; nt < 4; ++nt) { float y = acc[mt][nt][v]; s += y * y; }
        qacc[mt * 8 + v] += s;
      }
  }

#pragma unroll
  for (int i = 0; i < 16; ++i) {
    float s = qacc[i];
    s += __shfl_xor(s, 8, 32);
    s += __shfl_xor(s, 4, 32);
    s += __shfl_xor(s, 2, 32);
    s += __shfl_xor(s, 1, 32);
    qacc[i] = s;
  }
  if (lsub == 0) {
#pragma unroll
    for (int mt = 0; mt < 2; ++mt)
#pragma unroll
      for (int v = 0; v < 8; ++v)
        rowsum[wn][wm * 32 + mt * 16 + half * 8 + v] = qacc[mt * 8 + v];
  }
  __syncthreads();
  if (t < BM)
    ws[(size_t)blockIdx.y * BATCH + m0 + t] = rowsum[0][t] + rowsum[1][t];
}

// ---------------------------------------------------------------------------
// Kernel: out[b] = sum_g ws[g][b] + dot(x[b], q) + c   (one wave per row)
// ---------------------------------------------------------------------------
__global__ __launch_bounds__(256) void finalize_kernel(
    const float* __restrict__ x, const float* __restrict__ q,
    const float* __restrict__ c, const float* __restrict__ ws,
    float* __restrict__ out)
{
  const int lane = threadIdx.x & 31;
  const int wave = threadIdx.x >> 5;
  const int row  = blockIdx.x * 8 + wave;
  if (row >= BATCH) return;

  const float* xr = x + (size_t)row * DIM;
  float s = 0.f;
  for (int i = lane * 4; i < DIM; i += 128) {
    float4 xv = *(const float4*)(xr + i);
    float4 qv = *(const float4*)(q + i);
    s += xv.x * qv.x + xv.y * qv.y + xv.z * qv.z + xv.w * qv.w;
  }
  s += __shfl_xor(s, 16, 32);
  s += __shfl_xor(s, 8, 32);
  s += __shfl_xor(s, 4, 32);
  s += __shfl_xor(s, 2, 32);
  s += __shfl_xor(s, 1, 32);
  if (lane == 0) {
    float acc = s + c[0];
#pragma unroll
    for (int g = 0; g < NGROUP; ++g) acc += ws[(size_t)g * BATCH + row];
    out[row] = acc;
  }
}

// ---------------------------------------------------------------------------
extern "C" void kernel_launch(void* const* d_in, const int* in_sizes, int n_in,
                              void* d_out, int out_size, void* d_ws, size_t ws_size,
                              hipStream_t stream) {
  (void)in_sizes; (void)n_in; (void)out_size;
  const float* x  = (const float*)d_in[0];   // [BATCH, DIM]
  const float* sq = (const float*)d_in[1];   // [DIM, DIM]
  const float* q  = (const float*)d_in[2];   // [DIM]
  const float* c  = (const float*)d_in[3];   // [1]
  float* out = (float*)d_out;                // [BATCH]
  float* ws  = (float*)d_ws;

  const size_t partial_bytes = (size_t)NGROUP * BATCH * sizeof(float);   // 256 KB
  const size_t nx = (size_t)BATCH * DIM;     // 33.5M elems
  const size_t nq = (size_t)DIM * DIM;       // 16.8M elems
  const size_t plane_bytes = (nx + nq) * 2 * sizeof(__bf16);             // 192 MB
  dim3 grid(BATCH / BM, NGROUP);

  if (ws_size >= partial_bytes + plane_bytes) {
    // fast path: one-time precision split into workspace planes
    char* base = (char*)d_ws + partial_bytes;
    __bf16* xhi = (__bf16*)base;
    __bf16* xlo = xhi + nx;
    __bf16* qhi = xlo + nx;
    __bf16* qlo = qhi + nq;
    split_kernel<<<(int)(nx / (256 * 8)), 256, 0, stream>>>(x, xhi, xlo);
    split_kernel<<<(int)(nq / (256 * 8)), 256, 0, stream>>>(sq, qhi, qlo);
    quad_gemm_planes_kernel<<<grid, 256, 0, stream>>>(xhi, xlo, qhi, qlo, ws);
  } else {
    quad_gemm_kernel<<<grid, 256, 0, stream>>>(x, sq, ws);
  }
  finalize_kernel<<<BATCH / 8, 256, 0, stream>>>(x, q, c, ws, out);
}
